// MeanShift_9577777070267
// MI455X (gfx1250) — matrix operand order repaired and checked by reading, not compiled, not run on hardware
//
#include <hip/hip_runtime.h>

#define B_ 1024
#define K_ 128000
#define D_ 512
#define NCHUNK 64          // K chunks -> 2000 cols each
#define CH_COLS 2000
#define CH_TILES 125       // 2000 / 16
#define MROWS 64           // query rows per workgroup
#define NRB (B_ / MROWS)   // 16 row blocks
#define WAVES 4            // waves per workgroup in topk kernel
#define APAD 520           // padded LDS row stride (elems): 1040 B -> 4-bank skew

typedef __attribute__((ext_vector_type(16))) __bf16 v16bf;
typedef __attribute__((ext_vector_type(8)))  __bf16 v8bf;
typedef __attribute__((ext_vector_type(8)))  float  v8f;

__device__ __forceinline__ unsigned short f2bf(float x) {
    unsigned int u = __float_as_uint(x);
    unsigned int r = (u + 0x7fffu + ((u >> 16) & 1u)) >> 16;
    return (unsigned short)r;
}

__device__ __forceinline__ v16bf pack16(v8bf lo, v8bf hi) {
    v16bf v;
    #pragma unroll
    for (int q = 0; q < 8; ++q) { v[q] = lo[q]; v[q + 8] = hi[q]; }
    return v;
}

// ---------------- prep: tgt = l2norm(current_target), f32 + bf16 ----------------
__global__ void prep_tgt_kernel(const float* __restrict__ ct,
                                float* __restrict__ tgt,
                                unsigned short* __restrict__ tgtb) {
    int r = blockIdx.x, t = threadIdx.x;
    __shared__ float ss[256];
    float a = 0.f;
    for (int j = t; j < D_; j += 256) { float x = ct[r * D_ + j]; a += x * x; }
    ss[t] = a; __syncthreads();
    for (int w = 128; w > 0; w >>= 1) { if (t < w) ss[t] += ss[t + w]; __syncthreads(); }
    float inv = rsqrtf(ss[0]);
    for (int j = t; j < D_; j += 256) {
        float x = ct[r * D_ + j] * inv;
        tgt[r * D_ + j] = x;
        tgtb[r * D_ + j] = f2bf(x);
    }
}

// ---------------- prep: bank rows (bf16): tgt for r<B else l2norm(queue) --------
__global__ void prep_bank_kernel(const float* __restrict__ queue,
                                 const float* __restrict__ tgt,
                                 unsigned short* __restrict__ bank) {
    int r = blockIdx.x, t = threadIdx.x; // 128 threads
    __shared__ float ss[128];
    if (r < B_) {
        for (int j = t; j < D_; j += 128) bank[(size_t)r * D_ + j] = f2bf(tgt[r * D_ + j]);
    } else {
        float a = 0.f;
        for (int j = t; j < D_; j += 128) { float x = queue[(size_t)r * D_ + j]; a += x * x; }
        ss[t] = a; __syncthreads();
        for (int w = 64; w > 0; w >>= 1) { if (t < w) ss[t] += ss[t + w]; __syncthreads(); }
        float inv = rsqrtf(ss[0]);
        for (int j = t; j < D_; j += 128) bank[(size_t)r * D_ + j] = f2bf(queue[(size_t)r * D_ + j] * inv);
    }
}

// ---------------- tiny f32 GEMM: Y[i,j] = X[i,:]@W[:,j] + bias[j] ---------------
__global__ void gemm_bias_kernel(const float* __restrict__ X, const float* __restrict__ W,
                                 const float* __restrict__ bias, float* __restrict__ Y) {
    int i = blockIdx.x >> 1;
    int j = ((blockIdx.x & 1) << 8) + threadIdx.x;
    float s = bias[j];
    const float* xr = X + i * D_;
    #pragma unroll 4
    for (int k = 0; k < D_; ++k) s = fmaf(xr[k], W[k * D_ + j], s);
    Y[i * D_ + j] = s;
}

// ---------------- per-column batchnorm (training stats, biased var), in place ----
__global__ void bn_col_kernel(float* __restrict__ Z, const float* __restrict__ g,
                              const float* __restrict__ be, int do_relu) {
    int j = blockIdx.x, t = threadIdx.x;
    __shared__ float s1[256], s2[256];
    float a = 0.f, b = 0.f;
    for (int i = t; i < B_; i += 256) { float x = Z[i * D_ + j]; a += x; b += x * x; }
    s1[t] = a; s2[t] = b; __syncthreads();
    for (int w = 128; w > 0; w >>= 1) {
        if (t < w) { s1[t] += s1[t + w]; s2[t] += s2[t + w]; }
        __syncthreads();
    }
    float m  = s1[0] * (1.f / B_);
    float v  = s2[0] * (1.f / B_) - m * m;
    float iv = rsqrtf(v + 1e-5f);
    float gg = g[j], bb = be[j];
    for (int i = t; i < B_; i += 256) {
        float x = (Z[i * D_ + j] - m) * iv * gg + bb;
        if (do_relu) x = fmaxf(x, 0.f);
        Z[i * D_ + j] = x;
    }
}

// ---------------- row L2 normalize -> query ----------------
__global__ void l2row_kernel(const float* __restrict__ Z, float* __restrict__ Q) {
    int i = blockIdx.x, t = threadIdx.x;
    __shared__ float ss[256];
    float a = 0.f;
    for (int j = t; j < D_; j += 256) { float x = Z[i * D_ + j]; a += x * x; }
    ss[t] = a; __syncthreads();
    for (int w = 128; w > 0; w >>= 1) { if (t < w) ss[t] += ss[t + w]; __syncthreads(); }
    float inv = rsqrtf(ss[0]);
    for (int j = t; j < D_; j += 256) Q[i * D_ + j] = Z[i * D_ + j] * inv;
}

// ---------------- main WMMA kernel: streaming top-5 of tgt @ bank^T -------------
// 4 waves per workgroup share one LDS A-tile (64 rows); each wave owns one
// 2000-column chunk. Both the B fragment (global/VMEM) and the A fragment
// (LDS/DS) are double-buffered in registers so the fully-unrolled 64-step
// (ks x mt) pipeline always has the next fragment in flight.
__global__ __launch_bounds__(128)
void topk_wmma_kernel(const unsigned short* __restrict__ tgtb,
                      const unsigned short* __restrict__ bank,
                      float* __restrict__ cand_dot, int* __restrict__ cand_idx) {
    const int tid   = threadIdx.x;
    const int wave  = tid >> 5;
    const int lane  = tid & 31;
    const int rb    = blockIdx.x / (NCHUNK / WAVES);
    const int chgrp = blockIdx.x % (NCHUNK / WAVES);
    const int ch    = chgrp * WAVES + wave;
    const int mbase = rb * MROWS;
    const int nb0   = ch * CH_COLS;

    __shared__ __bf16 smA[MROWS * APAD];          // 66560 B, padded rows
    __shared__ float  cst[WAVES][4 * 16 * 16];    // 16 KB C staging, per wave

    // cooperative load of the A tile (64 x 512 bf16) into LDS
    {
        const __bf16* src = (const __bf16*)(tgtb + (size_t)mbase * D_);
        for (int c = tid; c < MROWS * (D_ / 8); c += 128) {
            int row = c >> 6, seg = c & 63;                 // 64 v8bf segments per row
            *(v8bf*)(smA + row * APAD + seg * 8) = *(const v8bf*)(src + row * D_ + seg * 8);
        }
    }
    __syncthreads();

    float dot5[2][5]; int idx5[2][5];
    #pragma unroll
    for (int r = 0; r < 2; ++r)
        #pragma unroll
        for (int j = 0; j < 5; ++j) { dot5[r][j] = -1e30f; idx5[r][j] = 0; }

    const int l15  = lane & 15;
    const int hi   = lane >> 4;
    const int selA = hi * 8;    // A-matrix 16-bit lane layout: K sub-offset
    const int selB = hi * 8;    // B-matrix 16-bit lane layout: K sub-offset
    const __bf16* aRowBase = smA + l15 * APAD + selA;

    for (int nt = 0; nt < CH_TILES; ++nt) {
        const int nbase = nb0 + nt * 16;
        const __bf16* bcol = (const __bf16*)(bank + (size_t)(nbase + l15) * D_ + selB);

        v8f acc[4];
        #pragma unroll
        for (int mt = 0; mt < 4; ++mt)
            #pragma unroll
            for (int j = 0; j < 8; ++j) acc[mt][j] = 0.f;

        // fragment double buffers (registers)
        v16bf bfrag[2];
        v16bf afrag[2];
        {
            v8bf blo = *(const v8bf*)(bcol);
            v8bf bhi = *(const v8bf*)(bcol + 16);
            bfrag[0] = pack16(blo, bhi);
            const __bf16* ap = aRowBase;                    // ks=0, mt=0
            v8bf alo = *(const v8bf*)(ap);
            v8bf ahi = *(const v8bf*)(ap + 16);
            afrag[0] = pack16(alo, ahi);
        }

        // flattened (ks, mt) pipeline: 16 k-steps x 4 M-tiles = 64 WMMAs
        #pragma unroll
        for (int i = 0; i < 64; ++i) {
            const int ks = i >> 2, mt = i & 3;
            const int cb = ks & 1;
            const int ca = i & 1;
            if (mt == 0 && ks < 15) {                       // prefetch next B fragment
                const __bf16* bp = bcol + (ks + 1) * 32;
                v8bf blo = *(const v8bf*)(bp);
                v8bf bhi = *(const v8bf*)(bp + 16);
                bfrag[cb ^ 1] = pack16(blo, bhi);
            }
            if (i < 63) {                                   // prefetch next A fragment
                const int ni = i + 1;
                const int nks = ni >> 2, nmt = ni & 3;
                const __bf16* ap = aRowBase + nmt * (16 * APAD) + nks * 32;
                v8bf alo = *(const v8bf*)(ap);
                v8bf ahi = *(const v8bf*)(ap + 16);
                afrag[ca ^ 1] = pack16(alo, ahi);
            }
            acc[mt] = __builtin_amdgcn_wmma_f32_16x16x32_bf16(
                false, afrag[ca], false, bfrag[cb], (short)0, acc[mt], false, false);
        }

        __syncthreads();
        // stage C tiles: lane(0-15): N=lane, M=j ; lane(16-31): N=lane-16, M=8+j
        #pragma unroll
        for (int mt = 0; mt < 4; ++mt)
            #pragma unroll
            for (int j = 0; j < 8; ++j)
                cst[wave][mt * 256 + (hi * 8 + j) * 16 + l15] = acc[mt][j];
        __syncthreads();

        // top-5 update (max dot): each lane owns local rows 2*lane, 2*lane+1
        #pragma unroll
        for (int r = 0; r < 2; ++r) {
            int lrow = 2 * lane + r;
            const float* rowp = &cst[wave][(lrow >> 4) * 256 + (lrow & 15) * 16];
            #pragma unroll
            for (int c = 0; c < 16; ++c) {
                float d = rowp[c];
                if (d > dot5[r][4]) {
                    dot5[r][4] = d; idx5[r][4] = nbase + c;
                    #pragma unroll
                    for (int j = 4; j > 0; --j) {
                        if (dot5[r][j] > dot5[r][j - 1]) {
                            float td = dot5[r][j]; dot5[r][j] = dot5[r][j - 1]; dot5[r][j - 1] = td;
                            int   ti = idx5[r][j]; idx5[r][j] = idx5[r][j - 1]; idx5[r][j - 1] = ti;
                        }
                    }
                }
            }
        }
    }

    #pragma unroll
    for (int r = 0; r < 2; ++r) {
        int row = mbase + 2 * lane + r;
        size_t base = ((size_t)row * NCHUNK + ch) * 5;
        #pragma unroll
        for (int j = 0; j < 5; ++j) { cand_dot[base + j] = dot5[r][j]; cand_idx[base + j] = idx5[r][j]; }
    }
}

// ---------------- merge per-chunk candidates -> final top-5 indices -------------
__global__ void merge_topk_kernel(const float* __restrict__ cand_dot,
                                  const int* __restrict__ cand_idx,
                                  int* __restrict__ nn_idx) {
    int row = blockIdx.x * blockDim.x + threadIdx.x;
    if (row >= B_) return;
    float d5[5]; int i5[5];
    #pragma unroll
    for (int j = 0; j < 5; ++j) { d5[j] = -1e30f; i5[j] = 0; }
    const float* cd = cand_dot + (size_t)row * (NCHUNK * 5);
    const int*   ci = cand_idx + (size_t)row * (NCHUNK * 5);
    for (int t = 0; t < NCHUNK * 5; ++t) {
        float d = cd[t];
        if (d > d5[4]) {
            d5[4] = d; i5[4] = ci[t];
            #pragma unroll
            for (int j = 4; j > 0; --j) {
                if (d5[j] > d5[j - 1]) {
                    float td = d5[j]; d5[j] = d5[j - 1]; d5[j - 1] = td;
                    int   ti = i5[j]; i5[j] = i5[j - 1]; i5[j - 1] = ti;
                }
            }
        }
    }
    #pragma unroll
    for (int j = 0; j < 5; ++j) nn_idx[row * 5 + j] = i5[j];
}

// ---------------- exact f32 dist_q at 5 neighbors + labels ----------------------
__global__ void finalize_kernel(const float* __restrict__ query, const float* __restrict__ tgt,
                                const float* __restrict__ queue, const int* __restrict__ labels,
                                const int* __restrict__ labels_q, const int* __restrict__ nn_idx,
                                float* __restrict__ rowLoss, float* __restrict__ rowPur) {
    int row = blockIdx.x, t = threadIdx.x; // 64 threads
    __shared__ float s1[64], s2[64];
    __shared__ float accL, accP;
    if (t == 0) { accL = 0.f; accP = 0.f; }
    const float* q = query + row * D_;
    for (int j = 0; j < 5; ++j) {
        int idx = nn_idx[row * 5 + j];
        const float* v = (idx < B_) ? (tgt + (size_t)idx * D_) : (queue + (size_t)idx * D_);
        float a = 0.f, b = 0.f;
        for (int k = t; k < D_; k += 64) { float x = v[k]; a += x * x; b += q[k] * x; }
        s1[t] = a; s2[t] = b; __syncthreads();
        for (int w = 32; w > 0; w >>= 1) {
            if (t < w) { s1[t] += s1[t + w]; s2[t] += s2[t + w]; }
            __syncthreads();
        }
        if (t == 0) {
            float dist = 2.f - 2.f * s2[0] * rsqrtf(s1[0]); // bank row is l2-normalized
            accL += dist;
            int lab = (idx < B_) ? labels[idx] : labels_q[idx];
            accP += (lab == labels[row]) ? 1.f : 0.f;
        }
        __syncthreads();
    }
    if (t == 0) { rowLoss[row] = accL * 0.2f; rowPur[row] = accP * 0.2f; }
}

// ---------------- deterministic final reduction ----------------------------------
__global__ void reduce_out_kernel(const float* __restrict__ rowLoss,
                                  const float* __restrict__ rowPur,
                                  float* __restrict__ out) {
    __shared__ float s1[256], s2[256];
    int t = threadIdx.x;
    float a = 0.f, b = 0.f;
    for (int i = t; i < B_; i += 256) { a += rowLoss[i]; b += rowPur[i]; }
    s1[t] = a; s2[t] = b; __syncthreads();
    for (int w = 128; w > 0; w >>= 1) {
        if (t < w) { s1[t] += s1[t + w]; s2[t] += s2[t + w]; }
        __syncthreads();
    }
    if (t == 0) { out[0] = s1[0] * (1.f / B_); out[1] = s2[0] * (1.f / B_); }
}

extern "C" void kernel_launch(void* const* d_in, const int* in_sizes, int n_in,
                              void* d_out, int out_size, void* d_ws, size_t ws_size,
                              hipStream_t stream) {
    const float* feat_q = (const float*)d_in[0];
    const float* ctgt   = (const float*)d_in[1];
    const float* queue  = (const float*)d_in[2];
    const float* W1     = (const float*)d_in[3];
    const float* b1     = (const float*)d_in[4];
    const float* g1     = (const float*)d_in[5];
    const float* be1    = (const float*)d_in[6];
    const float* W2     = (const float*)d_in[7];
    const float* b2     = (const float*)d_in[8];
    const float* g2     = (const float*)d_in[9];
    const float* be2    = (const float*)d_in[10];
    const int*   labels = (const int*)d_in[11];
    const int*   labq   = (const int*)d_in[12];
    float* out = (float*)d_out;

    char* ws = (char*)d_ws;
    size_t off = 0;
    float*          tgt      = (float*)(ws + off);           off += (size_t)B_ * D_ * 4;   // 2 MB
    unsigned short* tgtb     = (unsigned short*)(ws + off);  off += (size_t)B_ * D_ * 2;   // 1 MB
    unsigned short* bank     = (unsigned short*)(ws + off);  off += (size_t)K_ * D_ * 2;   // 131 MB
    float*          Z1       = (float*)(ws + off);           off += (size_t)B_ * D_ * 4;
    float*          Z2       = (float*)(ws + off);           off += (size_t)B_ * D_ * 4;
    float*          query    = (float*)(ws + off);           off += (size_t)B_ * D_ * 4;
    float*          cand_dot = (float*)(ws + off);           off += (size_t)B_ * NCHUNK * 5 * 4;
    int*            cand_idx = (int*)(ws + off);             off += (size_t)B_ * NCHUNK * 5 * 4;
    int*            nn_idx   = (int*)(ws + off);             off += (size_t)B_ * 5 * 4;
    float*          rowLoss  = (float*)(ws + off);           off += (size_t)B_ * 4;
    float*          rowPur   = (float*)(ws + off);           off += (size_t)B_ * 4;

    // target branch + memory bank (bf16)
    prep_tgt_kernel<<<B_, 256, 0, stream>>>(ctgt, tgt, tgtb);
    prep_bank_kernel<<<K_, 128, 0, stream>>>(queue, tgt, bank);

    // MLP: query = l2norm(bn(relu(bn(feat_q@W1+b1))@W2+b2))
    gemm_bias_kernel<<<B_ * 2, 256, 0, stream>>>(feat_q, W1, b1, Z1);
    bn_col_kernel<<<D_, 256, 0, stream>>>(Z1, g1, be1, 1);
    gemm_bias_kernel<<<B_ * 2, 256, 0, stream>>>(Z1, W2, b2, Z2);
    bn_col_kernel<<<D_, 256, 0, stream>>>(Z2, g2, be2, 0);
    l2row_kernel<<<B_, 256, 0, stream>>>(Z2, query);

    // big GEMM + streaming top-5 via bf16 WMMA (4 waves/WG share LDS A-tile)
    topk_wmma_kernel<<<NRB * (NCHUNK / WAVES), WAVES * 32, 0, stream>>>(tgtb, bank, cand_dot, cand_idx);
    merge_topk_kernel<<<(B_ + 255) / 256, 256, 0, stream>>>(cand_dot, cand_idx, nn_idx);

    // exact dist_q at selected neighbors, labels, final reduction
    finalize_kernel<<<B_, 64, 0, stream>>>(query, tgt, queue, labels, labq, nn_idx, rowLoss, rowPur);
    reduce_out_kernel<<<1, 256, 0, stream>>>(rowLoss, rowPur, out);
}
